// EdgeTransformerLayer_41068477284524
// MI455X (gfx1250) — matrix operand order repaired and checked by reading, not compiled, not run on hardware
//
#include <hip/hip_runtime.h>
#include <hip/hip_bf16.h>

// ---------------- problem constants ----------------
#define Bq   2
#define Nq   64
#define Dq   256
#define Hq   8
#define DKq  32
#define Rq   (Bq * Nq * Nq)          // 8192 rows of the [.,D] matrices
#define INV_SQRT_DK 0.1767766953f    // 1/sqrt(32)

typedef __attribute__((ext_vector_type(16))) __bf16 v16bf;
typedef __attribute__((ext_vector_type(8)))  float  v8f;
typedef __attribute__((ext_vector_type(2)))  float  v2f;
typedef __attribute__((ext_vector_type(4)))  unsigned int v4u;
typedef __attribute__((ext_vector_type(4)))  int    v4i;
typedef __attribute__((ext_vector_type(8)))  int    v8i;

#if __has_builtin(__builtin_amdgcn_tensor_load_to_lds)
#define HAVE_TDM 1
#endif

__device__ __forceinline__ unsigned short f2bf(float f) {
    unsigned u = __float_as_uint(f);
    unsigned r = u + 0x7FFFu + ((u >> 16) & 1u);   // round-to-nearest-even
    return (unsigned short)(r >> 16);
}

#ifdef HAVE_TDM
// Issue a TDM DMA: 2-D tile [rows x cols] of bf16 from global (row stride
// 'stride' elements) into LDS at byte offset lds_off, with 16B of LDS pad
// inserted after every 64B row (matches a 40-ushort padded pitch for 32 cols).
__device__ __forceinline__ void tdm_load_tile_bf16(
        const unsigned short* gptr, unsigned lds_off,
        unsigned cols, unsigned rows, unsigned stride,
        unsigned tensor_d0, unsigned tensor_d1) {
    unsigned long long ga = (unsigned long long)(const void*)gptr;
    v4u g0;
    g0[0] = 1u;                                       // count=1, user descriptor
    g0[1] = lds_off;                                  // D#.lds_addr (bytes)
    g0[2] = (unsigned)ga;                             // global_addr[31:0]
    g0[3] = (unsigned)((ga >> 32) & 0x01FFFFFFu) | (2u << 30);  // ga[56:32] | type=2
    v8i g1;
    // data_size=1 (2B), pad_enable, pad_interval=3 (16 DW = 64B), pad_amount=3 (4 DW = 16B)
    g1[0] = (int)((1u << 16) | (1u << 20) | (3u << 22) | (3u << 25));
    g1[1] = (int)((tensor_d0 & 0xFFFFu) << 16);                       // tensor_dim0[15:0]
    g1[2] = (int)((tensor_d0 >> 16) | ((tensor_d1 & 0xFFFFu) << 16)); // dim0 hi | dim1 lo
    g1[3] = (int)((tensor_d1 >> 16) | (cols << 16));                  // dim1 hi | tile_dim0
    g1[4] = (int)rows;                                                // tile_dim1 (tile_dim2=0)
    g1[5] = (int)stride;                                              // tensor_dim0_stride[31:0]
    g1[6] = 0; g1[7] = 0;
    v4i z4 = {0, 0, 0, 0};
#if defined(__clang_major__) && __clang_major__ >= 23
    v8i z8 = {0, 0, 0, 0, 0, 0, 0, 0};
    __builtin_amdgcn_tensor_load_to_lds(g0, g1, z4, z4, z8, 0);
#else
    __builtin_amdgcn_tensor_load_to_lds(g0, g1, z4, z4, 0);
#endif
}
#endif

// ---------------- conversion kernels ----------------
__global__ void cvt_bf16_kernel(unsigned short* __restrict__ dst,
                                const float* __restrict__ src, int n) {
    int i = blockIdx.x * 256 + threadIdx.x;
    if (i < n) dst[i] = f2bf(src[i]);
}

// Wcat[k, 1024] = [W_lk | W_rk | W_lv | W_rv][k, :]
__global__ void build_wcat_kernel(unsigned short* __restrict__ dst,
                                  const float* __restrict__ wlk,
                                  const float* __restrict__ wrk,
                                  const float* __restrict__ wlv,
                                  const float* __restrict__ wrv) {
    int i = blockIdx.x * 256 + threadIdx.x;     // i < 256*1024
    int k = i >> 10, np = i & 1023;
    int j = np >> 8, c = np & 255;
    const float* w = (j == 0) ? wlk : (j == 1) ? wrk : (j == 2) ? wlv : wrv;
    dst[i] = f2bf(w[k * Dq + c]);
}

// ---------------- generic bf16 WMMA GEMM ----------------
// C[M,Nn] = A[M,K] @ B[K,Nn] (+bias) (relu?) ; optional f32 and/or bf16 output.
// Tile: 64x64 per block (256 thr = 8 waves), BK=32, v_wmma_f32_16x16x32_bf16.
// A panel is DMA'd by the Tensor Data Mover (TENSORcnt), overlapped with the
// cooperative transposed staging of the B panel.
union FragBF { v16bf v; uint4 q[2]; };

__global__ __launch_bounds__(256)
void gemm_bf16_kernel(const unsigned short* __restrict__ A,
                      const unsigned short* __restrict__ Bm,
                      const float* __restrict__ bias,
                      float* __restrict__ Cf,
                      unsigned short* __restrict__ Cb,
                      int M, int Nn, int K, int relu) {
    __shared__ __align__(16) unsigned short As[64 * 40];   // [row][k], pad->40
    __shared__ __align__(16) unsigned short Bt[64 * 40];   // transposed: [col][k]

    const int t = threadIdx.x;
    const int m0 = blockIdx.y * 64;
    const int n0 = blockIdx.x * 64;

    const int w = t >> 5, lane = t & 31;
    const int wr = w >> 1;            // 0..3 : 16-row tile
    const int nbase = (w & 1) * 2;    // col tiles {nbase, nbase+1}
    const int mrow = 16 * wr + (lane & 15);
    const int kb  = (lane >> 4) * 8;   // A frag K-base (0 or 8)
    const int kb2 = (lane >> 4) * 16;  // unused; B frag also uses kb (0 or 8)

#ifdef HAVE_TDM
    const unsigned asOff = (unsigned)(size_t)(void*)&As[0];
#endif

    v8f acc0 = {}; v8f acc1 = {};

    for (int kt = 0; kt < K; kt += 32) {
#ifdef HAVE_TDM
        // --- A tile 64x32 via Tensor Data Mover (wave 0 issues; EXEC ignored) ---
        if (w == 0) {
            tdm_load_tile_bf16(A + (size_t)m0 * K + kt, asOff,
                               32u, 64u, (unsigned)K, (unsigned)K, (unsigned)M);
        }
#else
        {   // fallback: manual A staging
            int ar = t >> 2, ac = (t & 3) * 8;
            const unsigned short* gp = A + (size_t)(m0 + ar) * K + kt + ac;
            *(uint4*)&As[ar * 40 + ac] = *(const uint4*)gp;
        }
#endif
        // --- stage B tile 32x64, transposed into Bt[col][k] (overlaps the DMA) ---
        {
            int kr = t >> 3, nc0 = (t & 7) * 8;
            const unsigned short* gp = Bm + (size_t)(kt + kr) * Nn + n0 + nc0;
            union { uint4 q; unsigned short s[8]; } bv;
            bv.q = *(const uint4*)gp;
            if (kt + 32 < K) __builtin_prefetch(gp + (size_t)32 * Nn, 0, 0);
            #pragma unroll
            for (int i = 0; i < 8; ++i) Bt[(nc0 + i) * 40 + kr] = bv.s[i];
        }
#ifdef HAVE_TDM
        if (w == 0) __builtin_amdgcn_s_wait_tensorcnt(0);
#endif
        __syncthreads();

        FragBF af;
        af.q[0] = *(const uint4*)&As[mrow * 40 + kb];        // K = kb..kb+7
        af.q[1] = *(const uint4*)&As[mrow * 40 + kb + 16];   // K = kb+16..kb+23

        FragBF bf0, bf1;
        {
            int col = 16 * nbase + (lane & 15);
            bf0.q[0] = *(const uint4*)&Bt[col * 40 + kb];
            bf0.q[1] = *(const uint4*)&Bt[col * 40 + kb + 16];
            col += 16;
            bf1.q[0] = *(const uint4*)&Bt[col * 40 + kb];
            bf1.q[1] = *(const uint4*)&Bt[col * 40 + kb + 16];
        }
        acc0 = __builtin_amdgcn_wmma_f32_16x16x32_bf16(false, af.v, false, bf0.v,
                                                       (short)0, acc0, false, false);
        acc1 = __builtin_amdgcn_wmma_f32_16x16x32_bf16(false, af.v, false, bf1.v,
                                                       (short)0, acc1, false, false);
        __syncthreads();
    }

    // --- epilogue: C layout VGPR v -> M = v + (lane>=16 ? 8 : 0) ---
    #pragma unroll
    for (int nt = 0; nt < 2; ++nt) {
        const v8f& acc = nt ? acc1 : acc0;
        int coln = n0 + 16 * (nbase + nt) + (lane & 15);
        float bv = bias ? bias[coln] : 0.0f;
        #pragma unroll
        for (int v = 0; v < 8; ++v) {
            int row = m0 + 16 * wr + (lane >> 4) * 8 + v;
            float val = acc[v] + bv;
            if (relu) val = fmaxf(val, 0.0f);
            size_t idx = (size_t)row * Nn + coln;
            if (Cf) Cf[idx] = val;
            if (Cb) Cb[idx] = f2bf(val);
        }
    }
}

// ---------------- triangle score GEMM (exact fp32 WMMA) ----------------
// per block: one (b,h,a); S_a[x,y] = lk[b,x,a,h,:] . rk[b,a,y,h,:] / sqrt(DK)
__global__ __launch_bounds__(256)
void scores_kernel(const float* __restrict__ proj, float* __restrict__ S) {
    __shared__ __align__(16) float lkS[64 * 36];
    __shared__ __align__(16) float rkS[64 * 36];

    const int g = blockIdx.x;
    const int b = g >> 9, rem = g & 511;
    const int h = rem >> 6, a = rem & 63;
    const int t = threadIdx.x;

    {   // stage lk rows (x) and rk rows (y), 32 fp32 each, 8 per thread
        int r = t >> 2, c = (t & 3) * 8;
        const float* lp = proj + ((size_t)((b * 64 + r) * 64 + a)) * 1024 + h * 32 + c;
        *(float4*)&lkS[r * 36 + c]     = *(const float4*)lp;
        *(float4*)&lkS[r * 36 + c + 4] = *(const float4*)(lp + 4);
        const float* rp = proj + ((size_t)((b * 64 + a) * 64 + r)) * 1024 + 256 + h * 32 + c;
        *(float4*)&rkS[r * 36 + c]     = *(const float4*)rp;
        *(float4*)&rkS[r * 36 + c + 4] = *(const float4*)(rp + 4);
    }
    __syncthreads();

    const int w = t >> 5, lane = t & 31;
    const int wr = w >> 1, nbase = (w & 1) * 2;
    const int xl = 16 * wr + (lane & 15);
    const int y0 = 16 * nbase + (lane & 15);
    const int y1 = y0 + 16;

    v8f acc0 = {}; v8f acc1 = {};
    #pragma unroll
    for (int kk = 0; kk < 8; ++kk) {          // K = 32 in steps of 4
        int dk0 = kk * 4 + (lane >> 4) * 2;   // A 16x4 f32: lanes>=16 hold K=2,3
        v2f av; av[0] = lkS[xl * 36 + dk0]; av[1] = lkS[xl * 36 + dk0 + 1];
        v2f bv0; bv0[0] = rkS[y0 * 36 + dk0]; bv0[1] = rkS[y0 * 36 + dk0 + 1];
        v2f bv1; bv1[0] = rkS[y1 * 36 + dk0]; bv1[1] = rkS[y1 * 36 + dk0 + 1];
        acc0 = __builtin_amdgcn_wmma_f32_16x16x4_f32(false, av, false, bv0,
                                                     (short)0, acc0, false, false);
        acc1 = __builtin_amdgcn_wmma_f32_16x16x4_f32(false, av, false, bv1,
                                                     (short)0, acc1, false, false);
    }

    size_t sb = ((size_t)(b * 8 + h) * 64 + a) * 4096;  // S[b,h,a,x,y]
    #pragma unroll
    for (int nt = 0; nt < 2; ++nt) {
        const v8f& acc = nt ? acc1 : acc0;
        int y = 16 * (nbase + nt) + (lane & 15);
        #pragma unroll
        for (int v = 0; v < 8; ++v) {
            int x = 16 * wr + (lane >> 4) * 8 + v;
            S[sb + (size_t)x * 64 + y] = acc[v] * INV_SQRT_DK;
        }
    }
}

// ---------------- softmax over 'a' (axis 2), in place ----------------
__global__ __launch_bounds__(256)
void softmax_kernel(float* __restrict__ S) {
    int idx = blockIdx.x * 256 + threadIdx.x;          // < B*H*N*N
    int y = idx & 63, x = (idx >> 6) & 63;
    int h = (idx >> 12) & 7, b = idx >> 15;
    size_t base = ((size_t)(b * 8 + h) * 64) * 4096 + (size_t)x * 64 + y;
    const size_t st = 4096;
    float m = -3.4e38f;
    for (int a = 0; a < 64; ++a) m = fmaxf(m, S[base + a * st]);
    float sum = 0.0f;
    for (int a = 0; a < 64; ++a) { float e = __expf(S[base + a * st] - m); S[base + a * st] = e; sum += e; }
    float inv = 1.0f / sum;
    for (int a = 0; a < 64; ++a) S[base + a * st] *= inv;
}

// ---------------- triangle aggregation: o = sum_a att*lv*rv ----------------
// one wave per (b,h,x,y); lane = dk
__global__ __launch_bounds__(256)
void agg_kernel(const float* __restrict__ S, const float* __restrict__ proj,
                unsigned short* __restrict__ o_bf) {
    int wid = blockIdx.x * 8 + (threadIdx.x >> 5);
    int lane = threadIdx.x & 31;
    int y = wid & 63, x = (wid >> 6) & 63;
    int h = (wid >> 12) & 7, b = wid >> 15;

    size_t sbase = ((size_t)(b * 8 + h) * 64) * 4096 + (size_t)x * 64 + y;  // +a*4096
    const float* lv = proj + ((size_t)(b * 64 + x) * 64) * 1024 + 512 + h * 32 + lane;      // +a*1024
    const float* rv = proj + ((size_t)(b * 64) * 64 + y) * 1024 + 768 + h * 32 + lane;      // +a*65536

    float acc = 0.0f;
    for (int a = 0; a < 64; ++a)
        acc += S[sbase + (size_t)a * 4096] * lv[(size_t)a * 1024] * rv[(size_t)a * 65536];

    o_bf[((size_t)(b * 64 + x) * 64 + y) * 256 + h * 32 + lane] = f2bf(acc);
}

// ---------------- fused residual + LayerNorm (wave per row) ----------------
__global__ __launch_bounds__(256)
void ln_kernel(const float* __restrict__ r1, const float* __restrict__ r2,
               const float* __restrict__ g, const float* __restrict__ bta,
               float* __restrict__ out, unsigned short* __restrict__ outb) {
    int w = threadIdx.x >> 5, lane = threadIdx.x & 31;
    int row = blockIdx.x * 8 + w;
    const float* p1 = r1 + (size_t)row * 256;
    const float* p2 = r2 + (size_t)row * 256;
    float v[8], s = 0.0f, s2 = 0.0f;
    #pragma unroll
    for (int i = 0; i < 8; ++i) {
        int c = i * 32 + lane;
        float x = p1[c] + p2[c];
        v[i] = x; s += x; s2 += x * x;
    }
    #pragma unroll
    for (int off = 16; off > 0; off >>= 1) {
        s  += __shfl_xor(s,  off, 32);
        s2 += __shfl_xor(s2, off, 32);
    }
    float mean = s * (1.0f / 256.0f);
    float var  = s2 * (1.0f / 256.0f) - mean * mean;
    float inv  = rsqrtf(var + 1e-5f);
    #pragma unroll
    for (int i = 0; i < 8; ++i) {
        int c = i * 32 + lane;
        float o = (v[i] - mean) * inv * g[c] + bta[c];
        out[(size_t)row * 256 + c] = o;
        if (outb) outb[(size_t)row * 256 + c] = f2bf(o);
    }
}

// ---------------- launcher ----------------
extern "C" void kernel_launch(void* const* d_in, const int* in_sizes, int n_in,
                              void* d_out, int out_size, void* d_ws, size_t ws_size,
                              hipStream_t stream) {
    const float* x      = (const float*)d_in[0];
    const float* W_lk   = (const float*)d_in[1];
    const float* W_rk   = (const float*)d_in[2];
    const float* W_lv   = (const float*)d_in[3];
    const float* W_rv   = (const float*)d_in[4];
    const float* W_out  = (const float*)d_in[5];
    const float* ln1g   = (const float*)d_in[6];
    const float* ln1b   = (const float*)d_in[7];
    const float* W1     = (const float*)d_in[8];
    const float* b1     = (const float*)d_in[9];
    const float* W2     = (const float*)d_in[10];
    const float* b2     = (const float*)d_in[11];
    const float* ln2g   = (const float*)d_in[12];
    const float* ln2b   = (const float*)d_in[13];
    float* out = (float*)d_out;

    char* ws = (char*)d_ws;
    size_t off = 0;
    auto alloc = [&](size_t bytes) { size_t o = off; off = (off + bytes + 255) & ~(size_t)255; return o; };

    unsigned short* x_bf   = (unsigned short*)(ws + alloc((size_t)Rq * Dq * 2));
    unsigned short* wcat   = (unsigned short*)(ws + alloc((size_t)Dq * 4 * Dq * 2));
    unsigned short* wout_b = (unsigned short*)(ws + alloc((size_t)Dq * Dq * 2));
    unsigned short* w1_b   = (unsigned short*)(ws + alloc((size_t)Dq * 2 * Dq * 2));
    unsigned short* w2_b   = (unsigned short*)(ws + alloc((size_t)2 * Dq * Dq * 2));
    float*          proj   = (float*)(ws + alloc((size_t)Rq * 4 * Dq * 4));        // lk|rk|lv|rv
    float*          S      = (float*)(ws + alloc((size_t)Bq * Hq * Nq * Nq * Nq * 4));
    unsigned short* o_bf   = (unsigned short*)(ws + alloc((size_t)Rq * Dq * 2));
    float*          t1     = (float*)(ws + alloc((size_t)Rq * Dq * 4));            // o@Wout, then FFN2 out
    float*          h_f    = (float*)(ws + alloc((size_t)Rq * Dq * 4));
    unsigned short* h_bf   = (unsigned short*)(ws + alloc((size_t)Rq * Dq * 2));
    unsigned short* m1_bf  = (unsigned short*)(ws + alloc((size_t)Rq * 2 * Dq * 2));
    (void)ws_size; (void)in_sizes; (void)n_in; (void)out_size;

    // 1) bf16 conversions
    cvt_bf16_kernel<<<(Rq * Dq + 255) / 256, 256, 0, stream>>>(x_bf, x, Rq * Dq);
    build_wcat_kernel<<<(Dq * 4 * Dq + 255) / 256, 256, 0, stream>>>(wcat, W_lk, W_rk, W_lv, W_rv);
    cvt_bf16_kernel<<<(Dq * Dq + 255) / 256, 256, 0, stream>>>(wout_b, W_out, Dq * Dq);
    cvt_bf16_kernel<<<(Dq * 2 * Dq + 255) / 256, 256, 0, stream>>>(w1_b, W1, Dq * 2 * Dq);
    cvt_bf16_kernel<<<(2 * Dq * Dq + 255) / 256, 256, 0, stream>>>(w2_b, W2, 2 * Dq * Dq);

    // 2) fused projection GEMM: [8192,256] x [256,1024] -> proj (fp32)
    gemm_bf16_kernel<<<dim3(1024 / 64, Rq / 64), 256, 0, stream>>>(
        x_bf, wcat, nullptr, proj, nullptr, Rq, 1024, Dq, 0);

    // 3) triangle scores (exact f32 WMMA), then softmax over 'a'
    scores_kernel<<<Bq * Hq * Nq, 256, 0, stream>>>(proj, S);
    softmax_kernel<<<(Bq * Hq * Nq * Nq) / 256, 256, 0, stream>>>(S);

    // 4) triangle aggregation -> o (bf16)
    agg_kernel<<<(Bq * Hq * Nq * Nq) / 8, 256, 0, stream>>>(S, proj, o_bf);

    // 5) o @ W_out -> t1 ; h = LN(x + t1)
    gemm_bf16_kernel<<<dim3(Dq / 64, Rq / 64), 256, 0, stream>>>(
        o_bf, wout_b, nullptr, t1, nullptr, Rq, Dq, Dq, 0);
    ln_kernel<<<Rq / 8, 256, 0, stream>>>(x, t1, ln1g, ln1b, h_f, h_bf);

    // 6) FFN: relu(h@W1+b1)@W2+b2 ; out = LN(m + h)
    gemm_bf16_kernel<<<dim3(2 * Dq / 64, Rq / 64), 256, 0, stream>>>(
        h_bf, w1_b, b1, nullptr, m1_bf, Rq, 2 * Dq, Dq, 1);
    gemm_bf16_kernel<<<dim3(Dq / 64, Rq / 64), 256, 0, stream>>>(
        m1_bf, w2_b, b2, t1, nullptr, Rq, Dq, 2 * Dq, 0);
    ln_kernel<<<Rq / 8, 256, 0, stream>>>(h_f, t1, ln2g, ln2b, out, nullptr);
}
